// GeometricFrameAttention_25305947308326
// MI455X (gfx1250) — matrix operand order repaired and checked by reading, not compiled, not run on hardware
//
#include <hip/hip_runtime.h>
#include <hip/hip_bf16.h>
#include <math.h>

typedef __attribute__((ext_vector_type(16))) _Float16 v16h;
typedef __attribute__((ext_vector_type(8)))  float    v8f;

#define N_  512
#define H_  8
#define C_  128
#define CS_ 384
#define CZ_ 128
#define PQ_ 8
#define PV_ 8
#define INF_ 100000.0f
#define EPS_ 1e-8f

// ---------------- WMMA fragment helpers (contiguous-k A/B order, wave32) --------
// A 16x32 f16 fragment for one lane = two contiguous 8-element runs of row
// m = lane&15:  k in [kb, kb+8) and [kb+8, kb+16), kb = (lane>>4)*16.

__device__ inline v16h fragA_cvt(const float* __restrict__ row, int kb) {
  v16h f;
#pragma unroll
  for (int e = 0; e < 8; ++e) f[e] = (_Float16)row[kb + e];
#pragma unroll
  for (int e = 0; e < 8; ++e) f[8 + e] = (_Float16)row[kb + 8 + e];
  return f;
}

__device__ inline v16h fragA_cvt_scaled(const float* __restrict__ row, int kb, float sc) {
  v16h f;
#pragma unroll
  for (int e = 0; e < 8; ++e) f[e] = (_Float16)(row[kb + e] * sc);
#pragma unroll
  for (int e = 0; e < 8; ++e) f[8 + e] = (_Float16)(row[kb + 8 + e] * sc);
  return f;
}

// B 32x16 f16 fragment: lane covers K = kb..kb+15 (kb = (lane>>4)*16) at
// column n = lane&15; staged in LDS in fragment order -> one 32B v16h read.
// Writer mapping for element (kk,n): dstlane = n + 16*(kk>=16), elem = kk&15.
// C/D 16x16 f32: vgpr r -> row (r + 8*(lane>>4)), col lane&15.

// ---------------- quaternion helpers ----------------------------------------

__device__ inline void qrot(const float q[4], const float v[3], float o[3]) {
  float qw = q[0], qx = q[1], qy = q[2], qz = q[3];
  float uvx  = qy * v[2] - qz * v[1];
  float uvy  = qz * v[0] - qx * v[2];
  float uvz  = qx * v[1] - qy * v[0];
  float uuvx = qy * uvz - qz * uvy;
  float uuvy = qz * uvx - qx * uvz;
  float uuvz = qx * uvy - qy * uvx;
  o[0] = v[0] + 2.0f * (qw * uvx + uuvx);
  o[1] = v[1] + 2.0f * (qw * uvy + uuvy);
  o[2] = v[2] + 2.0f * (qw * uvz + uuvz);
}

__device__ inline void qmul(const float a[4], const float b[4], float o[4]) {
  o[0] = a[0]*b[0] - a[1]*b[1] - a[2]*b[2] - a[3]*b[3];
  o[1] = a[0]*b[1] + a[1]*b[0] + a[2]*b[3] - a[3]*b[2];
  o[2] = a[0]*b[2] - a[1]*b[3] + a[2]*b[0] + a[3]*b[1];
  o[3] = a[0]*b[3] + a[1]*b[2] - a[2]*b[1] + a[3]*b[0];
}

// ---------------- kernel 1: frames ------------------------------------------

__global__ void frames_k(const float* __restrict__ T, float* __restrict__ frames) {
  int n = blockIdx.x * blockDim.x + threadIdx.x;
  if (n >= N_) return;
  float q0 = T[n*16+0], q1 = T[n*16+1], q2 = T[n*16+2], q3 = T[n*16+3];
  float inv = 1.0f / sqrtf(q0*q0 + q1*q1 + q2*q2 + q3*q3 + EPS_);
  frames[n*8+0] = q0*inv; frames[n*8+1] = q1*inv;
  frames[n*8+2] = q2*inv; frames[n*8+3] = q3*inv;
  frames[n*8+4] = T[n*16+4]; frames[n*8+5] = T[n*16+5]; frames[n*8+6] = T[n*16+6];
  frames[n*8+7] = 0.0f;
}

// ---------------- kernel 2: generic WMMA GEMM (C = A@B + bias) --------------

__global__ void gemm16(const float* __restrict__ A, int lda,
                       const float* __restrict__ B, int ldb,
                       const float* __restrict__ bias,
                       float* __restrict__ C, int ldc, int K) {
  __shared__ __align__(32) _Float16 lBf[32 * 16];
  int lane = threadIdx.x;
  int m0 = blockIdx.y * 16;
  int n0 = blockIdx.x * 16;
  int kb = (lane >> 4) << 4;
  int dl = (lane >> 4) << 4;
  int el = lane & 15;
  const float* arow = A + (size_t)(m0 + (lane & 15)) * lda;
  float bv = bias ? bias[n0 + (lane & 15)] : 0.0f;
  v8f acc = {};
  for (int k0 = 0; k0 < K; k0 += 32) {
    const float* bp = B + (size_t)(k0 + lane) * ldb + n0;
#pragma unroll
    for (int t = 0; t < 16; ++t)
      lBf[(dl + t) * 16 + el] = (_Float16)bp[t];
    __syncthreads();
    v16h fa = fragA_cvt(arow + k0, kb);
    v16h fb = *(const v16h*)&lBf[lane * 16];
    acc = __builtin_amdgcn_wmma_f32_16x16x32_f16(false, fa, false, fb,
                                                 (short)0, acc, false, false);
    __syncthreads();
  }
  int n = lane & 15;
  int mb = (lane >> 4) * 8;
#pragma unroll
  for (int r = 0; r < 8; ++r)
    C[(size_t)(m0 + mb + r) * ldc + n0 + n] = acc[r] + bv;
}

// ---------------- kernel 3: fused single-pass over z ------------------------
// bias[i,j,h] = z[i,j,:]@W_b[:,h] ; dz[i,j,c] = z[i,j,:]@W_dz[:,c] + b_dz[c]
// z read exactly ONCE from HBM (134 MB, the roofline term). Each block does
// 4 row-tiles with one weight staging (4x less L2 weight traffic), and z
// tiles are double-buffered through LDS with async loads pipelined on the
// ASYNCcnt counter (issue next tile, wait <=4 so current tile is complete).

#define ZRT_ 4   // row-tiles per block

__device__ inline void async_tile(const float* __restrict__ z, size_t row0,
                                  int k0, unsigned ldsBase, int lane) {
#pragma unroll
  for (int it = 0; it < 4; ++it) {
    int cid = lane + 32 * it;            // 0..127 16B chunks of 16x32 f32 tile
    int r   = cid >> 3;
    int cb  = (cid & 7) * 16;
    unsigned long long ga =
        (unsigned long long)(uintptr_t)(z + (row0 + r) * CZ_ + k0) + cb;
    unsigned la = ldsBase + cid * 16;
    asm volatile("global_load_async_to_lds_b128 %0, %1, off"
                 :: "v"(la), "v"(ga) : "memory");
  }
}

__global__ void zpass_k(const float* __restrict__ z,
                        const float* __restrict__ W_b,
                        const float* __restrict__ W_dz,
                        const float* __restrict__ b_dz,
                        float* __restrict__ biasO,
                        float* __restrict__ dzO) {
  __shared__ __align__(16) float    lZ[2 * 16 * 32];      // double-buffered tile
  __shared__ __align__(32) _Float16 lWf[12 * 32 * 16];    // 4 ksteps x 3 coltiles
  int lane = threadIdx.x;

  // stage weights once per block: 128 x [8 W_b | 8 pad | 32 W_dz], frag order
  for (int t = 0; t < 192; ++t) {
    int idx = lane * 192 + t;
    int k = idx / 48, c = idx % 48;
    float v;
    if (c < 8)       v = W_b[k * 8 + c];
    else if (c < 16) v = 0.0f;
    else             v = W_dz[k * 32 + (c - 16)];
    int kstep = k >> 5, kk = k & 31;
    int ct = c >> 4,  n  = c & 15;
    int dstl = n + ((kk >> 4) << 4);
    lWf[(((kstep * 3 + ct) * 32) + dstl) * 16 + (kk & 15)] = (_Float16)v;
  }
  __syncthreads();

  unsigned lzBase = (unsigned)(uintptr_t)(void*)lZ;
  int kb = (lane >> 4) << 4;
  int n  = lane & 15;
  int mb = (lane >> 4) * 8;
  float bd0 = b_dz[n], bd1 = b_dz[16 + n];

  for (int rt = 0; rt < ZRT_; ++rt) {
    size_t row0 = ((size_t)blockIdx.x * ZRT_ + rt) * 16;
    v8f a0 = {}, a1 = {}, a2 = {};
    // prologue: first tile into buffer 0
    asm volatile("s_wait_dscnt 0x0" ::: "memory");
    async_tile(z, row0, 0, lzBase, lane);
    for (int kstep = 0; kstep < 4; ++kstep) {
      int cur = kstep & 1;
      if (kstep < 3) {
        asm volatile("s_wait_dscnt 0x0" ::: "memory");
        async_tile(z, row0, (kstep + 1) * 32, lzBase + (cur ^ 1) * 2048, lane);
        asm volatile("s_wait_asynccnt 0x4" ::: "memory"); // current tile done
      } else {
        asm volatile("s_wait_asynccnt 0x0" ::: "memory");
      }
      __syncthreads();
      const float* zrow = lZ + cur * 512 + (lane & 15) * 32;
      v16h fa = fragA_cvt(zrow, kb);
      v16h f0 = *(const v16h*)&lWf[((kstep * 3 + 0) * 32 + lane) * 16];
      v16h f1 = *(const v16h*)&lWf[((kstep * 3 + 1) * 32 + lane) * 16];
      v16h f2 = *(const v16h*)&lWf[((kstep * 3 + 2) * 32 + lane) * 16];
      a0 = __builtin_amdgcn_wmma_f32_16x16x32_f16(false, fa, false, f0, (short)0, a0, false, false);
      a1 = __builtin_amdgcn_wmma_f32_16x16x32_f16(false, fa, false, f1, (short)0, a1, false, false);
      a2 = __builtin_amdgcn_wmma_f32_16x16x32_f16(false, fa, false, f2, (short)0, a2, false, false);
      __syncthreads();
    }
#pragma unroll
    for (int r = 0; r < 8; ++r) {
      size_t row = row0 + mb + r;
      if (n < 8) biasO[row * 8 + n] = a0[r];
      dzO[row * 32 + n]      = a1[r] + bd0;
      dzO[row * 32 + 16 + n] = a2[r] + bd1;
    }
  }
}

// ---------------- kernel 4: geometry (points, v_g mix + transform) ----------

__global__ void geom_k(const float* __restrict__ frames,
                       const float* __restrict__ qpraw,
                       const float* __restrict__ kpraw,
                       const float* __restrict__ vgraw,
                       const float* __restrict__ g,
                       const float* __restrict__ W_mg,
                       float* __restrict__ qpf, float* __restrict__ kpf,
                       float* __restrict__ qsq, float* __restrict__ ksq,
                       float* __restrict__ vgglob) {
  int n = blockIdx.x;
  int t = threadIdx.x;           // hp = h*8+p
  int h = t >> 3, p = t & 7;
  __shared__ float ssq[64], ssk[64];
  float q[4] = {frames[n*8+0], frames[n*8+1], frames[n*8+2], frames[n*8+3]};
  float tr[3] = {frames[n*8+4], frames[n*8+5], frames[n*8+6]};

  float v[3], r[3];
#pragma unroll
  for (int d = 0; d < 3; ++d) v[d] = qpraw[n*192 + d*64 + t];
  qrot(q, v, r);
#pragma unroll
  for (int d = 0; d < 3; ++d) { r[d] += tr[d]; qpf[n*192 + h*24 + p*3 + d] = r[d]; }
  ssq[t] = r[0]*r[0] + r[1]*r[1] + r[2]*r[2];
#pragma unroll
  for (int d = 0; d < 3; ++d) v[d] = kpraw[n*192 + d*64 + t];
  qrot(q, v, r);
#pragma unroll
  for (int d = 0; d < 3; ++d) { r[d] += tr[d]; kpf[n*192 + h*24 + p*3 + d] = r[d]; }
  ssk[t] = r[0]*r[0] + r[1]*r[1] + r[2]*r[2];
  __syncthreads();
  if (p == 0) {
    float sq = 0.0f, sk = 0.0f;
#pragma unroll
    for (int pp = 0; pp < 8; ++pp) { sq += ssq[h*8+pp]; sk += ssk[h*8+pp]; }
    qsq[n*8+h] = sq; ksq[n*8+h] = sk;
  }
  float mv[16];
#pragma unroll
  for (int c = 0; c < 16; ++c) {
    float acc = 0.0f;
    for (int i = 0; i < 16; ++i) {
      float cv = (i < 8) ? vgraw[n*128 + i*16 + c] : g[n*128 + (i-8)*16 + c];
      acc += W_mg[t*16 + i] * cv;
    }
    mv[c] = acc;
  }
  float out[16];
  out[0] = mv[0];
  qrot(q, mv + 1, r); out[1] = r[0]; out[2] = r[1]; out[3] = r[2];
  qrot(q, mv + 4, r); out[4] = r[0]; out[5] = r[1]; out[6] = r[2];
  qrot(q, mv + 7, r);
  out[7] = r[0] + mv[10]*tr[0]; out[8] = r[1] + mv[10]*tr[1]; out[9] = r[2] + mv[10]*tr[2];
#pragma unroll
  for (int c = 10; c < 16; ++c) out[c] = mv[c];
#pragma unroll
  for (int c = 0; c < 16; ++c) vgglob[n*1024 + t*16 + c] = out[c];
}

// ---------------- kernel 4b: transpose k / k_pts for coalesced B staging ----
// kT[h][c][j]  (8x128x512), kpT[h][k][j] (8x24x512)

__global__ void ktrans_k(const float* __restrict__ kvbuf,
                         const float* __restrict__ kpf,
                         float* __restrict__ kT, float* __restrict__ kpT) {
  int idx = blockIdx.x * 256 + threadIdx.x;
  if (idx < 8 * 128 * 512) {
    int j = idx & 511, rem = idx >> 9;
    int c = rem & 127, h = rem >> 7;
    kT[idx] = kvbuf[(size_t)j * 2048 + h * 256 + c];
  }
  if (idx < 8 * 24 * 512) {
    int j = idx & 511, rem = idx >> 9;
    int k = rem % 24, h = rem / 24;
    kpT[idx] = kpf[(size_t)j * 192 + h * 24 + k];
  }
}

// ---------------- kernel 5: attention logits + softmax ----------------------
// grid = (N/16, H), blockDim = 32. A-fragments (q*s1, qp*hw) built once into
// registers; B tiles staged from transposed kT/kpT (contiguous 16-float runs).

__global__ void logits_k(const float* __restrict__ qbuf,
                         const float* __restrict__ kT,
                         const float* __restrict__ qpf,
                         const float* __restrict__ kpT,
                         const float* __restrict__ qsq,
                         const float* __restrict__ ksq,
                         const float* __restrict__ biasO,
                         const float* __restrict__ b_b,
                         const float* __restrict__ mask,
                         const float* __restrict__ head_weights,
                         const float* __restrict__ softmax_weights,
                         float* __restrict__ aOut) {
  __shared__ __align__(32) _Float16 lBf[32 * 16];
  __shared__ float logits[16 * 512];     // 32 KB
  int lane = threadIdx.x;
  int h  = blockIdx.y;
  int i0 = blockIdx.x * 16;

  const float s1 = 0.051031036307982884f;   // sqrt(1/(3C))
  const float s2 = 0.5773502691896258f;     // sqrt(1/3)
  float hw  = log1pf(expf(head_weights[h])) * 0.09622504486493764f; // softplus*sqrt(1/108)
  float smw = softmax_weights[h];
  float bb  = b_b[h];

  int kb = (lane >> 4) << 4;
  int dl = (lane >> 4) << 4;
  int el = lane & 15;

  // jt-invariant A fragments
  const float* qrow = qbuf + (size_t)(i0 + (lane & 15)) * 1024 + h * 128;
  v16h fq0 = fragA_cvt_scaled(qrow + 0,  kb, s1);
  v16h fq1 = fragA_cvt_scaled(qrow + 32, kb, s1);
  v16h fq2 = fragA_cvt_scaled(qrow + 64, kb, s1);
  v16h fq3 = fragA_cvt_scaled(qrow + 96, kb, s1);
  const float* prow = qpf + (size_t)(i0 + (lane & 15)) * 192 + h * 24;
  v16h fp;
#pragma unroll
  for (int e = 0; e < 8; ++e) {
    int k = kb + e;
    fp[e] = (k < 24) ? (_Float16)(prow[k] * hw) : (_Float16)0.0f;
  }
#pragma unroll
  for (int e = 0; e < 8; ++e) {
    int k = kb + 8 + e;
    fp[8 + e] = (k < 24) ? (_Float16)(prow[k] * hw) : (_Float16)0.0f;
  }
  int mbE = (lane >> 4) * 8;
  float qsI[8], mkI[8];
#pragma unroll
  for (int r = 0; r < 8; ++r) {
    int i = i0 + mbE + r;
    qsI[r] = qsq[i * 8 + h];
    mkI[r] = mask[i];
  }

  for (int jt = 0; jt < 32; ++jt) {
    int j0 = jt * 16;
    v8f acc = {};
#pragma unroll
    for (int ks = 0; ks < 4; ++ks) {             // q.k over C=128
      const float* bp = kT + ((size_t)h * 128 + ks * 32 + lane) * 512 + j0;
#pragma unroll
      for (int t = 0; t < 16; ++t)
        lBf[(dl + t) * 16 + el] = (_Float16)bp[t];
      __syncthreads();
      v16h fb = *(const v16h*)&lBf[lane * 16];
      v16h fa = (ks == 0) ? fq0 : (ks == 1) ? fq1 : (ks == 2) ? fq2 : fq3;
      acc = __builtin_amdgcn_wmma_f32_16x16x32_f16(false, fa, false, fb,
                                                   (short)0, acc, false, false);
      __syncthreads();
    }
    {                                            // point cross term, K=24 pad 32
      const float* bp = kpT + ((size_t)h * 24 + lane) * 512 + j0;
#pragma unroll
      for (int t = 0; t < 16; ++t) {
        float val = (lane < 24) ? bp[t] : 0.0f;
        lBf[(dl + t) * 16 + el] = (_Float16)val;
      }
      __syncthreads();
      v16h fb = *(const v16h*)&lBf[lane * 16];
      acc = __builtin_amdgcn_wmma_f32_16x16x32_f16(false, fp, false, fb,
                                                   (short)0, acc, false, false);
      __syncthreads();
    }
    int nn = lane & 15;
#pragma unroll
    for (int r = 0; r < 8; ++r) {
      int m = mbE + r, i = i0 + m, j = j0 + nn;
      float val = acc[r]
                + s2 * (biasO[((size_t)i * 512 + j) * 8 + h] + bb)
                - 0.5f * hw * (qsI[r] + ksq[j * 8 + h])
                + INF_ * (mkI[r] * mask[j] - 1.0f);
      logits[m * 512 + j] = smw * val;
    }
  }
  __syncthreads();
  if (lane < 16) {
    int r = lane;
    float mx = -3.0e38f;
    for (int j = 0; j < 512; ++j) mx = fmaxf(mx, logits[r * 512 + j]);
    float sum = 0.0f;
    for (int j = 0; j < 512; ++j) {
      float e = expf(logits[r * 512 + j] - mx);
      logits[r * 512 + j] = e;
      sum += e;
    }
    float inv = 1.0f / sum;
    for (int j = 0; j < 512; ++j)
      aOut[((size_t)h * 512 + i0 + r) * 512 + j] = logits[r * 512 + j] * inv;
  }
}

// ---------------- kernel 6: o_pair / o_rel (i-dependent batched GEMV) -------

__global__ void pair_rel_k(const float* __restrict__ aP,
                           const float* __restrict__ dzO,
                           const float* __restrict__ frames,
                           float* __restrict__ opair,
                           float* __restrict__ orel) {
  int i = blockIdx.x;
  int t = threadIdx.x;
  __shared__ float aRow[8 * 512];       // 16 KB
  for (int idx = t; idx < 4096; idx += 256) {
    int h = idx >> 9, j = idx & 511;
    aRow[idx] = aP[((size_t)h * 512 + i) * 512 + j];
  }
  __syncthreads();
  {
    int h = t >> 5, c = t & 31;
    float acc = 0.0f;
    for (int j = 0; j < 512; ++j)
      acc += aRow[h * 512 + j] * dzO[((size_t)i * 512 + j) * 32 + c];
    opair[(i * 8 + h) * 32 + c] = acc;
  }
  if (t < 64) {
    int h = t >> 3, c = t & 7;
    float qc[4] = {frames[i*8+0], -frames[i*8+1], -frames[i*8+2], -frames[i*8+3]};
    float ti[3] = {frames[i*8+4], frames[i*8+5], frames[i*8+6]};
    float acc = 0.0f;
    for (int j = 0; j < 512; ++j) {
      float qj[4] = {frames[j*8+0], frames[j*8+1], frames[j*8+2], frames[j*8+3]};
      float dtv[3] = {frames[j*8+4]-ti[0], frames[j*8+5]-ti[1], frames[j*8+6]-ti[2]};
      float rel[8];
      qmul(qc, qj, rel);
      float tr3[3]; qrot(qc, dtv, tr3);
      rel[4] = tr3[0]; rel[5] = tr3[1]; rel[6] = tr3[2]; rel[7] = 0.0f;
      acc += aRow[h * 512 + j] * rel[c];
    }
    orel[(i * 8 + h) * 8 + c] = acc;
  }
}

// ---------------- kernel 7: finalize (inv transform, norms, feats, g_out) ---

__global__ void finalize_k(const float* __restrict__ ogglob,
                           const float* __restrict__ obuf,
                           const float* __restrict__ opair,
                           const float* __restrict__ orel,
                           const float* __restrict__ frames,
                           const float* __restrict__ W_geo,
                           float* __restrict__ feats,
                           float* __restrict__ gout) {
  int i = blockIdx.x;
  int t = threadIdx.x;
  __shared__ float ogL[64 * 16];
  __shared__ float nrm[64 * 2];
  if (t < 64) {
    float qc[4] = {frames[i*8+0], -frames[i*8+1], -frames[i*8+2], -frames[i*8+3]};
    float tr[3] = {frames[i*8+4], frames[i*8+5], frames[i*8+6]};
    float in[16], out[16], r[3];
#pragma unroll
    for (int c = 0; c < 16; ++c) in[c] = ogglob[(size_t)i * 1024 + t * 16 + c];
    out[0] = in[0];
    qrot(qc, in + 1, r); out[1] = r[0]; out[2] = r[1]; out[3] = r[2];
    qrot(qc, in + 4, r); out[4] = r[0]; out[5] = r[1]; out[6] = r[2];
    float tmp[3] = {in[7]-in[10]*tr[0], in[8]-in[10]*tr[1], in[9]-in[10]*tr[2]};
    qrot(qc, tmp, r); out[7] = r[0]; out[8] = r[1]; out[9] = r[2];
#pragma unroll
    for (int c = 10; c < 16; ++c) out[c] = in[c];
    float n1 = 0.0f, n2 = 0.0f;
#pragma unroll
    for (int c = 0; c < 10; ++c) n1 += out[c]*out[c];
#pragma unroll
    for (int c = 10; c < 16; ++c) n2 += out[c]*out[c];
    nrm[t * 2 + 0] = sqrtf(n1 + EPS_);
    nrm[t * 2 + 1] = sqrtf(n2 + EPS_);
#pragma unroll
    for (int c = 0; c < 16; ++c) ogL[t * 16 + c] = out[c];
  }
  __syncthreads();
  for (int idx = t; idx < 2496; idx += 128) {
    int h = idx / 312, r = idx - h * 312;
    float v;
    if (r < 32)       v = opair[(i * 8 + h) * 32 + r];
    else if (r < 160) v = obuf[(size_t)i * 1024 + h * 128 + (r - 32)];
    else if (r < 168) v = orel[(i * 8 + h) * 8 + (r - 160)];
    else if (r < 296) v = ogL[h * 128 + (r - 168)];
    else              v = nrm[h * 16 + (r - 296)];
    feats[(size_t)i * 2496 + idx] = v;
  }
  if (t < 128) {
    int o = t >> 4, c = t & 15;
    float acc = 0.0f;
    for (int k = 0; k < 64; ++k) acc += W_geo[o * 64 + k] * ogL[k * 16 + c];
    gout[i * 128 + o * 16 + c] = acc;
  }
}

// ---------------- host driver ------------------------------------------------

extern "C" void kernel_launch(void* const* d_in, const int* in_sizes, int n_in,
                              void* d_out, int out_size, void* d_ws, size_t ws_size,
                              hipStream_t stream) {
  const float* s    = (const float*)d_in[0];
  const float* g    = (const float*)d_in[1];
  const float* z    = (const float*)d_in[2];
  const float* T    = (const float*)d_in[3];
  const float* mask = (const float*)d_in[4];
  const float* W_q  = (const float*)d_in[5];
  const float* b_q  = (const float*)d_in[6];
  const float* W_kv = (const float*)d_in[7];
  const float* b_kv = (const float*)d_in[8];
  const float* W_qp = (const float*)d_in[9];
  const float* b_qp = (const float*)d_in[10];
  const float* W_kp = (const float*)d_in[11];
  const float* b_kp = (const float*)d_in[12];
  const float* W_vg = (const float*)d_in[13];
  const float* b_vg = (const float*)d_in[14];
  const float* W_mg = (const float*)d_in[15];
  const float* W_b  = (const float*)d_in[16];
  const float* b_b  = (const float*)d_in[17];
  const float* W_dz = (const float*)d_in[18];
  const float* b_dz = (const float*)d_in[19];
  const float* head_weights    = (const float*)d_in[20];
  const float* softmax_weights = (const float*)d_in[21];
  const float* W_out = (const float*)d_in[22];
  const float* b_out = (const float*)d_in[23];
  const float* W_geo = (const float*)d_in[24];

  float* out = (float*)d_out;

  // workspace layout (floats)
  float* ws = (float*)d_ws;
  size_t off = 0;
  float* frames = ws + off; off += 512 * 8;
  float* qbuf   = ws + off; off += 512 * 1024;
  float* kvbuf  = ws + off; off += 512 * 2048;
  float* qpraw  = ws + off; off += 512 * 192;
  float* kpraw  = ws + off; off += 512 * 192;
  float* vgraw  = ws + off; off += 512 * 128;
  float* qpf    = ws + off; off += 512 * 192;
  float* kpf    = ws + off; off += 512 * 192;
  float* qsqb   = ws + off; off += 512 * 8;
  float* ksqb   = ws + off; off += 512 * 8;
  float* vgglob = ws + off; off += 512 * 1024;
  float* kT     = ws + off; off += (size_t)8 * 128 * 512;
  float* kpT    = ws + off; off += (size_t)8 * 24 * 512;
  float* biasO  = ws + off; off += (size_t)512 * 512 * 8;
  float* dzO    = ws + off; off += (size_t)512 * 512 * 32;
  float* aP     = ws + off; off += (size_t)8 * 512 * 512;
  float* obuf   = ws + off; off += 512 * 1024;
  float* ogglob = ws + off; off += 512 * 1024;
  float* opair  = ws + off; off += 512 * 8 * 32;
  float* orel   = ws + off; off += 512 * 8 * 8;
  float* feats  = ws + off; off += (size_t)512 * 2496;

  // 1) frames
  frames_k<<<2, 256, 0, stream>>>(T, frames);

  // 2) projections from s (WMMA GEMMs)
  gemm16<<<dim3(64, 32),  32, 0, stream>>>(s, CS_, W_q,  1024, b_q,  qbuf,  1024, CS_);
  gemm16<<<dim3(128, 32), 32, 0, stream>>>(s, CS_, W_kv, 2048, b_kv, kvbuf, 2048, CS_);
  gemm16<<<dim3(12, 32),  32, 0, stream>>>(s, CS_, W_qp, 192,  b_qp, qpraw, 192,  CS_);
  gemm16<<<dim3(12, 32),  32, 0, stream>>>(s, CS_, W_kp, 192,  b_kp, kpraw, 192,  CS_);
  gemm16<<<dim3(8, 32),   32, 0, stream>>>(s, CS_, W_vg, 128,  b_vg, vgraw, 128,  CS_);

  // 3) single fused async-pipelined pass over z (memory-roofline critical)
  zpass_k<<<(512 * 512) / (16 * ZRT_), 32, 0, stream>>>(z, W_b, W_dz, b_dz, biasO, dzO);

  // 4) geometry, then transpose k / k_pts for coalesced logits B staging
  geom_k<<<512, 64, 0, stream>>>(frames, qpraw, kpraw, vgraw, g, W_mg,
                                 qpf, kpf, qsqb, ksqb, vgglob);
  ktrans_k<<<(8 * 128 * 512) / 256, 256, 0, stream>>>(kvbuf, kpf, kT, kpT);

  // 5) attention logits + softmax
  logits_k<<<dim3(32, 8), 32, 0, stream>>>(qbuf, kT, qpf, kpT, qsqb, ksqb,
                                           biasO, b_b, mask, head_weights,
                                           softmax_weights, aP);

  // 6) attention-apply GEMMs per head: o = a@v ; o_g = a@vg_glob
  for (int h = 0; h < 8; ++h) {
    gemm16<<<dim3(8, 32), 32, 0, stream>>>(aP + (size_t)h * 512 * 512, 512,
                                           kvbuf + h * 256 + 128, 2048, nullptr,
                                           obuf + h * 128, 1024, 512);
    gemm16<<<dim3(8, 32), 32, 0, stream>>>(aP + (size_t)h * 512 * 512, 512,
                                           vgglob + h * 128, 1024, nullptr,
                                           ogglob + h * 128, 1024, 512);
  }

  // 7) o_pair / o_rel
  pair_rel_k<<<512, 256, 0, stream>>>(aP, dzO, frames, opair, orel);

  // 8) finalize: inverse transform, norms, feats assembly, g_out
  finalize_k<<<512, 128, 0, stream>>>(ogglob, obuf, opair, orel, frames, W_geo,
                                      feats, out + 512 * 384);

  // 9) s_out = feats @ W_out + b_out
  gemm16<<<dim3(24, 32), 32, 0, stream>>>(feats, 2496, W_out, CS_, b_out,
                                          out, CS_, 2496);
}
